// GConvCNP_54881092108482
// MI455X (gfx1250) — compile-verified, matrix operand order checked by reading, not run on hardware
//
#include <hip/hip_runtime.h>
#include <hip/hip_bf16.h>
#include <math.h>

// ---------------- problem constants (from reference) ----------------
#define B_      8
#define N_      2048
#define M_      2048
#define NUM_T_  268
#define T_      (NUM_T_ + M_)      // 2316 target points per batch
#define KNB     25                 // K_NBHD
#define KPAD    32                 // K padded to WMMA K=32
#define WNK     32                 // weightnet hidden width
#define CMCO_   16                 // weightnet output width
#define TILES_PER_B ((T_ + 15) / 16)   // 145

typedef __attribute__((ext_vector_type(16))) __bf16 v16bf;
typedef __attribute__((ext_vector_type(8)))  float  v8f;

static __device__ __forceinline__ __bf16 f2bf(float f) {
  unsigned u = __builtin_bit_cast(unsigned, f);
  u += 0x7FFFu + ((u >> 16) & 1u);           // round-to-nearest-even
  unsigned short s = (unsigned short)(u >> 16);
  return __builtin_bit_cast(__bf16, s);
}
// fast sigmoid: v_exp_f32 + v_rcp_f32, no IEEE-div fixup chains
static __device__ __forceinline__ float sigm(float x) {
  return __builtin_amdgcn_rcpf(1.0f + __expf(-x));
}
static __device__ __forceinline__ float swish(float x) { return x * sigm(x); }

// K-index pattern for 16-bit WMMA A/B fragments (8 VGPRs, 2 bf16 each):
// lanes 0-15 cover K={0..7,16..23}, lanes 16-31 cover K={8..15,24..31}.
static __device__ __forceinline__ int kidx(int j, int half) {
  int base = (j < 4) ? (2 * j) : (16 + 2 * (j - 4));
  return base + 8 * half;
}

// ---------------- kernel 1: RBF smoother + fused pre-linear ----------------
__global__ void kmat_pre_kernel(const float* __restrict__ tc, const float* __restrict__ cc,
                                const float* __restrict__ cv,
                                const float* __restrict__ psi_os, const float* __restrict__ psi_ls,
                                const float* __restrict__ pre_w, const float* __restrict__ pre_b,
                                float* __restrict__ vals0) {
  int bt = blockIdx.x;                 // one block per (b, t)
  int b = bt / T_, t = bt % T_;
  float tcv = tc[b * T_ + t];
  float os  = psi_os[0];
  float ls  = psi_ls[0];
  float inv = -0.5f / (ls * ls);
  float s0 = 0.f, s1 = 0.f;
  for (int n = threadIdx.x; n < N_; n += blockDim.x) {
    float d = tcv - cc[b * N_ + n];
    float k = os * __expf(inv * d * d);
    s0 += k;
    s1 += k * cv[b * N_ + n];
  }
  __shared__ float r0[256], r1[256];
  r0[threadIdx.x] = s0; r1[threadIdx.x] = s1;
  __syncthreads();
  for (int s = 128; s > 0; s >>= 1) {
    if (threadIdx.x < s) { r0[threadIdx.x] += r0[threadIdx.x + s]; r1[threadIdx.x] += r1[threadIdx.x + s]; }
    __syncthreads();
  }
  if (threadIdx.x < 8) {
    float h0 = r0[0];
    float h1 = r1[0] / (h0 + 1e-8f);
    int c = threadIdx.x;
    float a = tcv * pre_w[0 * 8 + c] + h0 * pre_w[1 * 8 + c] + h1 * pre_w[2 * 8 + c] + pre_b[c];
    vals0[(long)(b * T_ + t) * 8 + c] = 1.0f / (1.0f + __expf(-a));  // exact-ish at source
  }
}

// ---------------- kernel 2: 25-NN in 1-D (shared by all 4 conv layers) ----------------
__global__ void topk_kernel(const float* __restrict__ tc, int* __restrict__ idx,
                            float* __restrict__ pairv) {
  int gid = blockIdx.x * blockDim.x + threadIdx.x;
  if (gid >= B_ * T_) return;
  int b = gid / T_, t = gid % T_;
  const float* c = tc + b * T_;
  float me = c[t];
  float bd[KNB]; int bi[KNB];
#pragma unroll
  for (int k = 0; k < KNB; k++) { bd[k] = 3.4e38f; bi[k] = 0; }
  float worst = 3.4e38f; int wslot = 0;
  for (int j = 0; j < T_; j++) {
    if ((j & 63) == 0) __builtin_prefetch(&c[j + 128], 0, 1);  // global_prefetch_b8
    float d = fabsf(me - c[j]);
    if (d < worst) {
      bd[wslot] = d; bi[wslot] = j;
      worst = -1.f;
#pragma unroll
      for (int k = 0; k < KNB; k++) if (bd[k] > worst) { worst = bd[k]; wslot = k; }
    }
  }
  for (int k = 0; k < KNB; k++) {
    idx[(long)gid * KNB + k]   = bi[k];
    pairv[(long)gid * KNB + k] = me - c[bi[k]];    // pairs[b,t,idx] = tc[t]-tc[idx]
  }
}

// ---------------- kernel 3: one LieConv layer, WMMA everywhere ----------------
// One wave (32 threads) per 16-point tile. 400 weightnet rows = 25 M-tiles of 16.
template <int CI, int CO>
__global__ void __launch_bounds__(32)
conv_kernel(const float* __restrict__ vals_in, float* __restrict__ vals_out,
            const int* __restrict__ idx, const float* __restrict__ pairv,
            const float* __restrict__ w1, const float* __restrict__ b1,
            const float* __restrict__ w2, const float* __restrict__ b2,
            const float* __restrict__ w3, const float* __restrict__ b3,
            const float* __restrict__ lw, const float* __restrict__ lb) {
  constexpr int CIP  = (CI < 16) ? 16 : CI;       // pad channel dim to WMMA M
  constexpr int ROWS = 16 * KNB;                  // 400 (point,neighbor) rows
  static_assert(ROWS % 16 == 0, "row tiles");

  __shared__ __bf16 sH [ROWS][WNK];               // h1, then (in-place) swish(h2)
  __shared__ __bf16 sKW[ROWS][CMCO_];             // weightnet output
  __shared__ __bf16 sNvT[16][CIP][KPAD];          // gathered nb_vals, transposed, K-padded
  __shared__ __bf16 sPC[16][CIP][CMCO_];          // per-point pc tiles
  __shared__ float  sW1[WNK], sB1[WNK], sB2[WNK], sB3[CMCO_];

  const int lane = threadIdx.x;
  const int tile = blockIdx.x;
  const int b    = tile / TILES_PER_B;
  const int t0   = (tile % TILES_PER_B) * 16;
  const int nloc = lane & 15, half = lane >> 4;

  if (lane < WNK) { sW1[lane] = w1[lane]; sB1[lane] = b1[lane]; sB2[lane] = b2[lane]; }
  if (lane < CMCO_) sB3[lane] = b3[lane];

  // ---- persistent B-fragments for w2 (32x32 -> two N-halves) and w3 (32x16) ----
  v16bf bw2[2]; v16bf bw3;
#pragma unroll
  for (int j = 0; j < 8; j++) {
    int k = kidx(j, half);
#pragma unroll
    for (int nh = 0; nh < 2; nh++) {
      bw2[nh][2 * j]     = f2bf(w2[k * WNK + (nloc + 16 * nh)]);
      bw2[nh][2 * j + 1] = f2bf(w2[(k + 1) * WNK + (nloc + 16 * nh)]);
    }
    bw3[2 * j]     = f2bf(w3[k * CMCO_ + nloc]);
    bw3[2 * j + 1] = f2bf(w3[(k + 1) * CMCO_ + nloc]);
  }

  // zero the K-padded gather buffer
  for (int e = lane; e < 16 * CIP * KPAD; e += 32) ((__bf16*)sNvT)[e] = f2bf(0.f);
  __syncthreads();

  // ---- gather pairs/neighbor values + weightnet layer 1 (1->32, swish) ----
  for (int r = lane; r < ROWS; r += 32) {
    int p = r / KNB, k = r % KNB;
    int t = t0 + p;
    float a = 0.f;
    if (t < T_) {
      long base = (long)(b * T_ + t) * KNB + k;
      a = pairv[base];
      int j = idx[base];
      const float* vp = vals_in + (long)(b * T_ + j) * CI;
#pragma unroll
      for (int i = 0; i < CI; i++) sNvT[p][i][k] = f2bf(vp[i]);
    }
#pragma unroll
    for (int j = 0; j < WNK; j++) sH[r][j] = f2bf(swish(a * sW1[j] + sB1[j]));
  }
  __syncthreads();

  // ---- weightnet layer 2: 32->32, 25 M-tiles x 2 N-halves, in-place update ----
  for (int m = 0; m < KNB; m++) {
    int R = m * 16;
    v16bf A;
#pragma unroll
    for (int j = 0; j < 8; j++) {
      int k = kidx(j, half);
      A[2 * j] = sH[R + nloc][k]; A[2 * j + 1] = sH[R + nloc][k + 1];
    }
#pragma unroll
    for (int nh = 0; nh < 2; nh++) {
      v8f c = {};
      c = __builtin_amdgcn_wmma_f32_16x16x32_bf16(false, A, false, bw2[nh], (short)0, c, false, false);
#pragma unroll
      for (int v = 0; v < 8; v++) {
        int me = v + 8 * half, o = nloc + 16 * nh;
        sH[R + me][o] = f2bf(swish(c[v] + sB2[o]));   // rows just consumed -> safe in-place
      }
    }
  }
  __syncthreads();

  // ---- weightnet layer 3: 32->16, 25 M-tiles ----
  for (int m = 0; m < KNB; m++) {
    int R = m * 16;
    v16bf A;
#pragma unroll
    for (int j = 0; j < 8; j++) {
      int k = kidx(j, half);
      A[2 * j] = sH[R + nloc][k]; A[2 * j + 1] = sH[R + nloc][k + 1];
    }
    v8f c = {};
    c = __builtin_amdgcn_wmma_f32_16x16x32_bf16(false, A, false, bw3, (short)0, c, false, false);
#pragma unroll
    for (int v = 0; v < 8; v++) {
      int me = v + 8 * half;
      sKW[R + me][nloc] = f2bf(swish(c[v] + sB3[nloc]));
    }
  }
  __syncthreads();

  // ---- einsum btki,btko->btio : per point, (CI x 25) x (25 x 16), K padded to 32 ----
  for (int p = 0; p < 16; p++) {
    v16bf Bf;
#pragma unroll
    for (int j = 0; j < 8; j++) {
      int k = kidx(j, half);
      Bf[2 * j]     = (k     < KNB) ? sKW[p * KNB + k][nloc]     : f2bf(0.f);
      Bf[2 * j + 1] = (k + 1 < KNB) ? sKW[p * KNB + k + 1][nloc] : f2bf(0.f);
    }
#pragma unroll
    for (int mt = 0; mt < CIP / 16; mt++) {
      v16bf A;
#pragma unroll
      for (int j = 0; j < 8; j++) {
        int k = kidx(j, half);
        A[2 * j]     = sNvT[p][mt * 16 + nloc][k];
        A[2 * j + 1] = sNvT[p][mt * 16 + nloc][k + 1];
      }
      v8f c = {};
      c = __builtin_amdgcn_wmma_f32_16x16x32_bf16(false, A, false, Bf, (short)0, c, false, false);
#pragma unroll
      for (int v = 0; v < 8; v++) sPC[p][mt * 16 + v + 8 * half][nloc] = f2bf(c[v]);
    }
  }
  __syncthreads();

  // ---- linear mix: M=16 points, K=CI*16, N=CO (accumulated WMMA chain) ----
  constexpr int KT = (CI * 16) / KPAD;
  constexpr int NT = (CO + 15) / 16;
#pragma unroll
  for (int nt = 0; nt < NT; nt++) {
    v8f c = {};
    for (int kt = 0; kt < KT; kt++) {
      v16bf A, Bf;
#pragma unroll
      for (int j = 0; j < 8; j++) {
        int k  = kidx(j, half);
        int g0 = kt * KPAD + k, g1 = g0 + 1;
        A[2 * j]     = sPC[nloc][g0 >> 4][g0 & 15];   // A rows = points
        A[2 * j + 1] = sPC[nloc][g1 >> 4][g1 & 15];
        int cg = nloc + 16 * nt;
        Bf[2 * j]     = (cg < CO) ? f2bf(lw[g0 * CO + cg]) : f2bf(0.f);
        Bf[2 * j + 1] = (cg < CO) ? f2bf(lw[g1 * CO + cg]) : f2bf(0.f);
      }
      c = __builtin_amdgcn_wmma_f32_16x16x32_bf16(false, A, false, Bf, (short)0, c, false, false);
    }
#pragma unroll
    for (int v = 0; v < 8; v++) {
      int p = v + 8 * half;
      int t = t0 + p;
      int cg = nloc + 16 * nt;
      if (t < T_ && cg < CO) {
        float o = (c[v] + lb[cg]) * (1.0f / KNB);
        vals_out[(long)(b * T_ + t) * CO + cg] = fmaxf(o, 0.f);   // relu from _forward
      }
    }
  }
}

// ---------------- head kernels ----------------
__global__ void mean_kernel(const float* __restrict__ vals, const float* __restrict__ mw,
                            const float* __restrict__ mb, float* __restrict__ out) {
  int gid = blockIdx.x * blockDim.x + threadIdx.x;
  if (gid >= B_ * M_) return;
  int b = gid / M_, m = gid % M_;
  const float* f = vals + (long)(b * T_ + NUM_T_ + m) * 8;
  float s = mb[0];
#pragma unroll
  for (int c = 0; c < 8; c++) s += f[c] * mw[c];
  out[gid] = s;
}

// var: (B, M, M) filled with 1e-8, diagonal = max(softplus(f@vw+vb), 1e-8).
// float4 stores -> GLOBAL_STORE_B128, 4x fewer store instructions on the 134MB write.
__global__ void var_kernel(const float* __restrict__ vals, const float* __restrict__ vw,
                           const float* __restrict__ vb, float* __restrict__ out) {
  long gid = (long)blockIdx.x * blockDim.x + threadIdx.x;     // one float4 per thread
  long total4 = (long)B_ * M_ * M_ / 4;
  if (gid >= total4) return;
  long e4  = gid * 4;
  int  b   = (int)(e4 / ((long)M_ * M_));
  long rem = e4 - (long)b * M_ * M_;
  int  i   = (int)(rem / M_);
  int  j0  = (int)(rem % M_);                                 // row-aligned: M % 4 == 0
  float4 v = make_float4(1e-8f, 1e-8f, 1e-8f, 1e-8f);
  if (i >= j0 && i < j0 + 4) {
    const float* f = vals + (long)(b * T_ + NUM_T_ + i) * 8;
    float s = vb[0];
#pragma unroll
    for (int c = 0; c < 8; c++) s += f[c] * vw[c];
    float sp = (s > 20.f) ? s : log1pf(__expf(s));            // softplus
    float d  = fmaxf(sp, 1e-8f);
    if (i == j0)     v.x = d;
    else if (i == j0 + 1) v.y = d;
    else if (i == j0 + 2) v.z = d;
    else                  v.w = d;
  }
  *(float4*)(out + e4) = v;
}

// ---------------- launch ----------------
// Input order assumption: jax-pytree (sorted dict key) flattening of setup_inputs():
//   0 ctx_coords, 1 ctx_values, 2 num_t,
//   3.. params sorted: convs[0..3]{b1,b2,b3,lin_b,lin_w,w1,w2,w3},
//   35 mean_b, 36 mean_w, 37 pre_b, 38 pre_w, 39 psi_ls, 40 psi_os, 41 var_b, 42 var_w,
//   43 t_coords
extern "C" void kernel_launch(void* const* d_in, const int* in_sizes, int n_in,
                              void* d_out, int out_size, void* d_ws, size_t ws_size,
                              hipStream_t stream) {
  const float* ctx_coords = (const float*)d_in[0];
  const float* ctx_values = (const float*)d_in[1];
  struct ConvP { const float *b1, *b2, *b3, *lb, *lw, *w1, *w2, *w3; } cp[4];
  for (int l = 0; l < 4; l++) {
    int base = 3 + l * 8;
    cp[l].b1 = (const float*)d_in[base + 0];
    cp[l].b2 = (const float*)d_in[base + 1];
    cp[l].b3 = (const float*)d_in[base + 2];
    cp[l].lb = (const float*)d_in[base + 3];
    cp[l].lw = (const float*)d_in[base + 4];
    cp[l].w1 = (const float*)d_in[base + 5];
    cp[l].w2 = (const float*)d_in[base + 6];
    cp[l].w3 = (const float*)d_in[base + 7];
  }
  const float* mean_b  = (const float*)d_in[35];
  const float* mean_w  = (const float*)d_in[36];
  const float* pre_b   = (const float*)d_in[37];
  const float* pre_w   = (const float*)d_in[38];
  const float* psi_ls  = (const float*)d_in[39];
  const float* psi_os  = (const float*)d_in[40];
  const float* var_b   = (const float*)d_in[41];
  const float* var_w   = (const float*)d_in[42];
  const float* t_coords = (const float*)d_in[43];

  char* ws = (char*)d_ws;
  float* valsA = (float*)ws;  ws += sizeof(float) * (size_t)B_ * T_ * 32;
  float* valsB = (float*)ws;  ws += sizeof(float) * (size_t)B_ * T_ * 32;
  int*   nidx  = (int*)ws;    ws += sizeof(int)   * (size_t)B_ * T_ * KNB;
  float* pairv = (float*)ws;

  kmat_pre_kernel<<<B_ * T_, 256, 0, stream>>>(t_coords, ctx_coords, ctx_values,
                                               psi_os, psi_ls, pre_w, pre_b, valsA);
  topk_kernel<<<(B_ * T_ + 255) / 256, 256, 0, stream>>>(t_coords, nidx, pairv);

  dim3 cg(B_ * TILES_PER_B);
  conv_kernel<8, 16><<<cg, 32, 0, stream>>>(valsA, valsB, nidx, pairv,
      cp[0].w1, cp[0].b1, cp[0].w2, cp[0].b2, cp[0].w3, cp[0].b3, cp[0].lw, cp[0].lb);
  conv_kernel<16, 32><<<cg, 32, 0, stream>>>(valsB, valsA, nidx, pairv,
      cp[1].w1, cp[1].b1, cp[1].w2, cp[1].b2, cp[1].w3, cp[1].b3, cp[1].lw, cp[1].lb);
  conv_kernel<32, 16><<<cg, 32, 0, stream>>>(valsA, valsB, nidx, pairv,
      cp[2].w1, cp[2].b1, cp[2].w2, cp[2].b2, cp[2].w3, cp[2].b3, cp[2].lw, cp[2].lb);
  conv_kernel<16, 8><<<cg, 32, 0, stream>>>(valsB, valsA, nidx, pairv,
      cp[3].w1, cp[3].b1, cp[3].w2, cp[3].b2, cp[3].w3, cp[3].b3, cp[3].lw, cp[3].lb);

  mean_kernel<<<(B_ * M_ + 255) / 256, 256, 0, stream>>>(valsA, mean_w, mean_b, (float*)d_out);
  long tot4 = (long)B_ * M_ * M_ / 4;
  var_kernel<<<(int)((tot4 + 255) / 256), 256, 0, stream>>>(valsA, var_w, var_b,
                                                            (float*)d_out + (size_t)B_ * M_);
}
